// KnowledgeFeatureFusion_53051436040668
// MI455X (gfx1250) — matrix operand order repaired and checked by reading, not compiled, not run on hardware
//
#include <hip/hip_runtime.h>
#include <hip/hip_bf16.h>

typedef __attribute__((ext_vector_type(16))) _Float16 v16h;
typedef __attribute__((ext_vector_type(8)))  float    v8f;

#define F_IN 8
#define H    32
#define FO   128
#define LO   64
#define KDIM 64   // K of both big GEMMs
#define MT   64   // M rows per block in the WMMA GEMM

// ---------------------------------------------------------------------------
// small helper kernels (GCN path)
// ---------------------------------------------------------------------------
__global__ void kff_zero(float* __restrict__ p, int n) {
    int i = blockIdx.x * blockDim.x + threadIdx.x;
    if (i < n) p[i] = 0.0f;
}

__global__ void kff_edge_deg(const int* __restrict__ dst, float* __restrict__ deg, int nE) {
    int e = blockIdx.x * blockDim.x + threadIdx.x;
    if (e < nE) atomicAdd(&deg[dst[e]], 1.0f);
}

__global__ void kff_dinv(float* __restrict__ deg, int n) {
    int i = blockIdx.x * blockDim.x + threadIdx.x;
    if (i < n) deg[i] = rsqrtf(deg[i] + 1.0f);   // in-place: deg -> dinv
}

// h_lin[M,32] = X[M,K] @ W[K,32]   (K = 8 or 32; one thread per node row)
__global__ void kff_lin(const float* __restrict__ X, const float* __restrict__ W,
                        float* __restrict__ h_lin, int M, int K) {
    int i = blockIdx.x * blockDim.x + threadIdx.x;
    if (i >= M) return;
    float x[32];
    for (int k = 0; k < K; ++k) x[k] = X[i * K + k];
    for (int c = 0; c < H; ++c) {
        float s = 0.0f;
        for (int k = 0; k < K; ++k) s += x[k] * W[k * H + c];
        h_lin[i * H + c] = s;
    }
}

// agg[dst] += h_lin[src] * dinv[src]*dinv[dst]   (one thread per edge)
__global__ void kff_agg(const int* __restrict__ src, const int* __restrict__ dst,
                        const float* __restrict__ h_lin, const float* __restrict__ dinv,
                        float* __restrict__ agg, int nE) {
    int e = blockIdx.x * blockDim.x + threadIdx.x;
    if (e >= nE) return;
    int s = src[e], d = dst[e];
    float c = dinv[s] * dinv[d];
    const float* hs = h_lin + (size_t)s * H;
    float* ad = agg + (size_t)d * H;
    for (int f = 0; f < H; ++f) atomicAdd(&ad[f], hs[f] * c);
}

// h_out = relu(agg + h_lin*dinv^2 + b); optionally accumulate sum into pooled[32]
__global__ __launch_bounds__(256)
void kff_post(const float* __restrict__ agg, const float* __restrict__ h_lin,
              const float* __restrict__ dinv, const float* __restrict__ b,
              float* __restrict__ h_out, float* __restrict__ pooled, int M) {
    __shared__ float ps[H];
    int t = blockIdx.x * blockDim.x + threadIdx.x;
    if (pooled) {
        if (threadIdx.x < H) ps[threadIdx.x] = 0.0f;
        __syncthreads();
    }
    if (t < M * H) {
        int i = t / H, f = t % H;
        float dv = dinv[i];
        float v = agg[t] + h_lin[t] * dv * dv + b[f];
        v = v > 0.0f ? v : 0.0f;
        if (h_out) h_out[t] = v;
        if (pooled) atomicAdd(&ps[f], v);
    }
    if (pooled) {
        __syncthreads();
        if (threadIdx.x < H) atomicAdd(&pooled[threadIdx.x], ps[threadIdx.x]);
    }
}

// on_vec[c] = (pooled/N) @ Wf_on + bf   (128 threads)
__global__ void kff_onvec(const float* __restrict__ pooled, const float* __restrict__ Wf_on,
                          const float* __restrict__ bf, float* __restrict__ on_vec,
                          float invN) {
    int c = threadIdx.x;
    float s = bf[c];
    for (int k = 0; k < H; ++k) s += pooled[k] * invN * Wf_on[k * FO + c];
    on_vec[c] = s;
}

// ---------------------------------------------------------------------------
// WMMA GEMM:  out[M,128] = A[M,64] @ Bw[64,128]  (+ fused epilogue)
// block = 256 threads = 8 waves; wave -> one 16-col slab, held B in regs;
// block covers MT=64 rows (4 sub-tiles of 16); grid.x = M/64
// ---------------------------------------------------------------------------
__global__ __launch_bounds__(256)
void kff_gemm_wmma(const float* __restrict__ A,       // M x 64 row-major f32
                   const float* __restrict__ Bw,      // 64 x 128 row-major f32
                   const float* __restrict__ hi_lin,  // (M/64) x 128 (fused only)
                   const float* __restrict__ on_vec,  // 128        (fused only)
                   const float* __restrict__ alpha_p, // 1          (fused only)
                   float* __restrict__ out,           // M x 128
                   int fused) {
    __shared__ _Float16 Atile[MT * KDIM];             // 64 rows x 64 K, f16 (8KB)

    const int tid = threadIdx.x;
    const int m0  = blockIdx.x * MT;

    // cooperative load of A tile: coalesced float4 -> packed f16 -> LDS
    const float4* A4 = reinterpret_cast<const float4*>(A + (size_t)m0 * KDIM);
#pragma unroll
    for (int it = 0; it < (MT * KDIM / 4) / 256; ++it) {
        int idx = it * 256 + tid;                     // float4 index
        float4 v = A4[idx];
        _Float16* dp = &Atile[idx * 4];
        dp[0] = (_Float16)v.x; dp[1] = (_Float16)v.y;
        dp[2] = (_Float16)v.z; dp[3] = (_Float16)v.w;
    }

    const int wave     = tid >> 5;        // 0..7 -> column slab
    const int lane     = tid & 31;
    const int nsub     = lane & 15;       // N within slab / A sub-row
    const int hi_half  = lane >> 4;       // 0: lanes 0-15, 1: lanes 16-31
    const int col_base = wave * 16;
    const int col      = col_base + nsub;

    // B regs (per wave, loaded ONCE, reused across 4 M sub-tiles):
    // elems 0..7 -> K kc*32 + hi_half*8 + i, elems 8..15 -> K kc*32 + 16 + hi_half*8 + (i-8) ; N = col
    v16h b[2];
#pragma unroll
    for (int kc = 0; kc < 2; ++kc) {
        const int kB = kc * 32 + hi_half * 8;
#pragma unroll
        for (int i = 0; i < 16; ++i)
            b[kc][i] = (_Float16)Bw[(size_t)(kB + i + (i >= 8 ? 8 : 0)) * FO + col];
    }

    // fused-epilogue operands: constant over the whole 64-row block
    float al = 0.25f, addv = 0.0f;
    if (fused) {
        al   = alpha_p[0];
        addv = on_vec[col] + hi_lin[(size_t)(m0 >> 6) * FO + col];
    }

    __syncthreads();

#pragma unroll
    for (int mt = 0; mt < MT / 16; ++mt) {
        const int row = mt * 16 + nsub;
        v8f acc = {};
#pragma unroll
        for (int kc = 0; kc < 2; ++kc) {
            // A regs: elems 0..7 -> K kA..kA+7, elems 8..15 -> K kA+16..kA+23
            const int kA = kc * 32 + hi_half * 8;
            v16h a;
#pragma unroll
            for (int i = 0; i < 8; ++i) {
                a[i]     = Atile[row * KDIM + kA + i];
                a[i + 8] = Atile[row * KDIM + kA + 16 + i];
            }
            acc = __builtin_amdgcn_wmma_f32_16x16x32_f16(
                /*neg_a=*/false, a, /*neg_b=*/false, b[kc],
                /*c_mod=*/(short)0, acc, /*reuse_a=*/false, /*reuse_b=*/false);
        }
        // C/D layout: VGPR j, lanes 0-15 -> row j; lanes 16-31 -> row j+8
#pragma unroll
        for (int j = 0; j < 8; ++j) {
            int m = m0 + mt * 16 + (hi_half ? j + 8 : j);
            float z = acc[j];
            if (fused) {
                z += addv;
                z = z >= 0.0f ? z : al * z;
            }
            out[(size_t)m * FO + col] = z;
        }
    }
}

// ---------------------------------------------------------------------------
extern "C" void kernel_launch(void* const* d_in, const int* in_sizes, int n_in,
                              void* d_out, int out_size, void* d_ws, size_t ws_size,
                              hipStream_t stream) {
    const float* high  = (const float*)d_in[0];   // (B,P,T,64)
    const float* low   = (const float*)d_in[1];   // (B,P,T,64,64)
    const float* xn    = (const float*)d_in[2];   // (N_NODES, 8)
    const int*   eidx  = (const int*)  d_in[3];   // (2, E)
    const float* W1    = (const float*)d_in[4];
    const float* b1    = (const float*)d_in[5];
    const float* W2    = (const float*)d_in[6];
    const float* b2    = (const float*)d_in[7];
    const float* Wf    = (const float*)d_in[8];   // (160,128)
    const float* bf    = (const float*)d_in[9];
    const float* alpha = (const float*)d_in[10];
    float* out = (float*)d_out;

    const int nNodes = in_sizes[2] / F_IN;
    const int nE     = in_sizes[3] / 2;
    const int Mhi    = in_sizes[0] / LO;          // 4096 rows of high
    const int Mlow   = in_sizes[1] / LO;          // 262144 rows of low
    const int M32    = nNodes * H;

    const int* src = eidx;
    const int* dst = eidx + nE;

    // workspace layout (floats)
    float* ws     = (float*)d_ws;
    float* deg    = ws;                    // nNodes  (becomes dinv)
    float* linb   = deg    + nNodes;       // M32  h1_lin, then h2_lin
    float* agg    = linb   + M32;          // M32  (reused for both layers)
    float* h1     = agg    + M32;          // M32
    float* pooled = h1     + M32;          // 32
    float* onv    = pooled + H;            // 128
    float* hilin  = onv    + FO;           // Mhi*128

    const int T = 256;
    auto g = [&](int n) { return (n + T - 1) / T; };

    // --- GCN layer 1 ---
    kff_zero<<<g(nNodes), T, 0, stream>>>(deg, nNodes);
    kff_zero<<<g(M32),    T, 0, stream>>>(agg, M32);
    kff_zero<<<1,         H, 0, stream>>>(pooled, H);
    kff_edge_deg<<<g(nE), T, 0, stream>>>(dst, deg, nE);
    kff_dinv<<<g(nNodes), T, 0, stream>>>(deg, nNodes);
    kff_lin<<<g(nNodes),  T, 0, stream>>>(xn, W1, linb, nNodes, F_IN);
    kff_agg<<<g(nE),      T, 0, stream>>>(src, dst, linb, deg, agg, nE);
    kff_post<<<g(M32),    T, 0, stream>>>(agg, linb, deg, b1, h1, nullptr, nNodes);

    // --- GCN layer 2 (+ mean-pool accumulation) ---
    kff_zero<<<g(M32),    T, 0, stream>>>(agg, M32);
    kff_lin<<<g(nNodes),  T, 0, stream>>>(h1, W2, linb, nNodes, H);
    kff_agg<<<g(nE),      T, 0, stream>>>(src, dst, linb, deg, agg, nE);
    kff_post<<<g(M32),    T, 0, stream>>>(agg, linb, deg, b2, nullptr, pooled, nNodes);

    // --- on_vec = (pooled/N) @ Wf_on + bf ---
    kff_onvec<<<1, FO, 0, stream>>>(pooled, Wf + (size_t)(LO + LO) * FO, bf, onv,
                                    1.0f / (float)nNodes);

    // --- hi_lin = high @ Wf_hi  (plain WMMA GEMM) ---
    kff_gemm_wmma<<<Mhi / MT, 256, 0, stream>>>(
        high, Wf + (size_t)LO * FO, nullptr, nullptr, nullptr, hilin, 0);

    // --- main fused GEMM: out = leaky(low @ Wf_lo + hi_lin + on_vec) ---
    kff_gemm_wmma<<<Mlow / MT, 256, 0, stream>>>(
        low, Wf, hilin, onv, alpha, out, 1);
}
